// LightweightSelfAttention_27419071218206
// MI455X (gfx1250) — matrix operand order repaired and checked by reading, not compiled, not run on hardware
//
#include <hip/hip_runtime.h>

typedef __attribute__((ext_vector_type(16))) _Float16 v16h;
typedef __attribute__((ext_vector_type(8)))  _Float16 v8h;
typedef __attribute__((ext_vector_type(2)))  _Float16 h2;
typedef __attribute__((ext_vector_type(8)))  float    v8f;
typedef __attribute__((ext_vector_type(4)))  float    v4f;

#define L_SEQ   2048
#define D_DIM   128
#define QT      128      // queries per workgroup
#define KT      64       // keys per iteration
#define NT      (L_SEQ / KT)
#define SQ_STR  136      // sQ row stride (halves): 128 + 8 pad, 16B-aligned rows
#define SK_STR  136      // sK row stride (halves)
#define SV_STR  72       // sVt row stride (halves): 64 + 8 pad
#define SP_STR  72       // sP row stride (halves)
// (1/sqrt(128)) * log2(e): scores scaled straight into the base-2 exp domain
#define K2SCALE 0.1275129566791819f

// A-matrix fragment (16x32 f16): lane%16 = row M; per ISA 7.12.2 the halves
// cover K as two contiguous 8-half runs: [kbase+g8, +8) and [kbase+16+g8, +8).
// Both runs are 16B aligned -> 2x ds_load_b128.
__device__ __forceinline__ v16h frag_a(const _Float16* p, int kbase, int g8) {
  v8h lo = *(const v8h*)(p + kbase + g8);
  v8h hi = *(const v8h*)(p + kbase + 16 + g8);
  return __builtin_shufflevector(lo, hi, 0, 1, 2, 3, 4, 5, 6, 7,
                                 8, 9, 10, 11, 12, 13, 14, 15);
}

// B-matrix fragment (32x16 f16): lane%16 = col N; halves cover K as two
// 8-half runs [kbase+(lane/16)*8, +8) and [kbase+16+(lane/16)*8, +8) -> 2x ds_load_b128.
__device__ __forceinline__ v16h frag_b(const _Float16* p, int kbase, int g16) {
  v8h lo = *(const v8h*)(p + kbase + (g16 >> 1));
  v8h hi = *(const v8h*)(p + kbase + 16 + (g16 >> 1));
  return __builtin_shufflevector(lo, hi, 0, 1, 2, 3, 4, 5, 6, 7,
                                 8, 9, 10, 11, 12, 13, 14, 15);
}

// Issue the 8 float4 global loads for one 64-key tile (kept in flight; the
// matching s_wait_loadcnt lands at the store phase of the *next* iteration).
__device__ __forceinline__ void ktile_load(v4f* buf, const float* __restrict__ xh,
                                           int k0, int tid) {
#pragma unroll
  for (int i = 0; i < 8; ++i) {
    const int e  = tid + 256 * i;
    const int d  = e >> 4;
    const int lp = (e & 15) << 2;
    buf[i] = *(const v4f*)(xh + d * L_SEQ + k0 + lp);
  }
}

// One workgroup = one (head, query-tile of 128 rows). 8 waves, each owns 16
// query rows. Flash-attention over 2048 keys in 64-key tiles with register
// double-buffered global staging; fused column-sum (mean over L) partial out.
__global__ __launch_bounds__(256) void attn_partial_kernel(const float* __restrict__ x,
                                                           float* __restrict__ ws) {
  extern __shared__ char smem_raw[];
  _Float16* sQ   = (_Float16*)smem_raw;          // [QT][SQ_STR]  queries, row-major [m][d]
  _Float16* sK   = sQ  + QT * SQ_STR;            // [KT][SK_STR]  keys, row-major [kk][d]
  _Float16* sVt  = sK  + KT * SK_STR;            // [D][SV_STR]   values, [d][kk] (== global layout)
  _Float16* sP   = sVt + D_DIM * SV_STR;         // [QT][SP_STR]  probs, row-major [m][kk]
  float*    sRed = (float*)(sP + QT * SP_STR);   // [8][128] cross-wave reduction

  const int tid  = threadIdx.x;
  const int lane = tid & 31;
  const int wv   = tid >> 5;       // wave 0..7
  const int ln16 = lane & 15;
  const int g8   = (lane >> 4) << 3;
  const int g16  = (lane >> 4) << 4;

  const int wg   = blockIdx.x;     // head*16 + qtile
  const int head = wg >> 4;
  const int qt   = wg & 15;
  const int q0   = qt * QT;
  const float* __restrict__ xh = x + (size_t)head * (D_DIM * L_SEQ);

  // ---- Q tile: stage all 16 float4 per thread in registers, then scatter ----
  v4f qbuf[16];
#pragma unroll
  for (int i = 0; i < 16; ++i) {
    const int e  = tid + 256 * i;
    const int d  = e >> 5;
    const int lp = (e & 31) << 2;
    qbuf[i] = *(const v4f*)(xh + d * L_SEQ + q0 + lp);
  }
#pragma unroll
  for (int i = 0; i < 16; ++i) {
    const int e  = tid + 256 * i;
    const int d  = e >> 5;
    const int lp = (e & 31) << 2;
#pragma unroll
    for (int j = 0; j < 4; ++j)
      sQ[(lp + j) * SQ_STR + d] = (_Float16)qbuf[i][j];
  }
  __syncthreads();

  // Persistent GEMM1 A-fragments: this wave's 16 query rows, all of D=128.
  const _Float16* qrow = sQ + (wv * 16 + ln16) * SQ_STR;
  v16h aQ[4];
#pragma unroll
  for (int ks = 0; ks < 4; ++ks) aQ[ks] = frag_a(qrow, 32 * ks, g8);

  const v8f vzero = {};
  v8f acc[8];                       // O accumulator: 16 rows x 128 dims
#pragma unroll
  for (int dt = 0; dt < 8; ++dt) acc[dt] = vzero;
  float m_run[8], s_run[8];         // per-row running max (log2 domain) / sum
#pragma unroll
  for (int r = 0; r < 8; ++r) { m_run[r] = -3.0e38f; s_run[r] = 0.0f; }

  // Prologue: tile 0 loads in flight.
  v4f buf[8];
  ktile_load(buf, xh, 0, tid);

  for (int kt = 0; kt < NT; ++kt) {
    __syncthreads();  // previous iteration done reading sK/sVt
    // Store staged tile to both LDS layouts (f32 -> f16).
#pragma unroll
    for (int i = 0; i < 8; ++i) {
      const int e  = tid + 256 * i;
      const int d  = e >> 4;
      const int lp = (e & 15) << 2;
      const _Float16 h0 = (_Float16)buf[i][0];
      const _Float16 h1 = (_Float16)buf[i][1];
      const _Float16 hh2 = (_Float16)buf[i][2];
      const _Float16 h3 = (_Float16)buf[i][3];
      h2 p01 = {h0, h1}, p23 = {hh2, h3};
      *(h2*)(sVt + d * SV_STR + lp)     = p01;   // [d][key] contiguous
      *(h2*)(sVt + d * SV_STR + lp + 2) = p23;
      sK[(lp + 0) * SK_STR + d] = h0;            // [key][d] transpose scatter
      sK[(lp + 1) * SK_STR + d] = h1;
      sK[(lp + 2) * SK_STR + d] = hh2;
      sK[(lp + 3) * SK_STR + d] = h3;
    }
    __syncthreads();
    // Prefetch next tile into registers; overlaps with the GEMMs below.
    if (kt + 1 < NT) ktile_load(buf, xh, (kt + 1) * KT, tid);

    // ---- GEMM1: S(16 x 64) = Q_rows x K_tile^T, one continuous 16-step
    // pipeline with one-fragment lookahead (partial dscnt waits throughout) ----
    v8f s[4];
#pragma unroll
    for (int ct = 0; ct < 4; ++ct) s[ct] = vzero;
    {
      v16h bk = frag_b(sK + ln16 * SK_STR, 0, g16);
#pragma unroll
      for (int idx = 0; idx < 16; ++idx) {
        const int ct = idx >> 2;
        const int ks = idx & 3;
        v16h bn;
        if (idx < 15) {
          const int ct2 = (idx + 1) >> 2;
          const int ks2 = (idx + 1) & 3;
          bn = frag_b(sK + (ct2 * 16 + ln16) * SK_STR, 32 * ks2, g16);
        }
        s[ct] = __builtin_amdgcn_wmma_f32_16x16x32_f16(false, aQ[ks], false, bk,
                                                       (short)0, s[ct], false, false);
        bk = bn;
      }
    }

    // ---- Online softmax (base-2 domain, scale folded into K2SCALE) ----
#pragma unroll
    for (int r = 0; r < 8; ++r) {
      float v = s[0][r];
#pragma unroll
      for (int ct = 1; ct < 4; ++ct) v = fmaxf(v, s[ct][r]);
#pragma unroll
      for (int off = 1; off < 16; off <<= 1) v = fmaxf(v, __shfl_xor(v, off, 32));
      v *= K2SCALE;
      const float m_new = fmaxf(m_run[r], v);
      const float fac   = __builtin_amdgcn_exp2f(m_run[r] - m_new);
      m_run[r] = m_new;
      s_run[r] *= fac;
#pragma unroll
      for (int dt = 0; dt < 8; ++dt) acc[dt][r] *= fac;
    }

    // P = 2^(S*K2 - m); accumulate row sums; spill f16 probs to own LDS rows.
    float rs[8];
#pragma unroll
    for (int r = 0; r < 8; ++r) rs[r] = 0.0f;
#pragma unroll
    for (int ct = 0; ct < 4; ++ct) {
#pragma unroll
      for (int r = 0; r < 8; ++r) {
        const float p = __builtin_amdgcn_exp2f(s[ct][r] * K2SCALE - m_run[r]);
        rs[r] += p;
        sP[(wv * 16 + r + g8) * SP_STR + (ct * 16 + ln16)] = (_Float16)p;
      }
    }
#pragma unroll
    for (int r = 0; r < 8; ++r) {
      float v = rs[r];
#pragma unroll
      for (int off = 1; off < 16; off <<= 1) v += __shfl_xor(v, off, 32);
      s_run[r] += v;
    }

    // ---- GEMM2: acc(16 x 128) += P(16 x 64) x V(64 x 128), flattened
    // 16-step pipeline, both A-fragments preloaded ----
    {
      const _Float16* prow = sP + (wv * 16 + ln16) * SP_STR;
      v16h aP[2];
      aP[0] = frag_a(prow, 0, g8);
      aP[1] = frag_a(prow, 32, g8);
      v16h bv = frag_b(sVt + ln16 * SV_STR, 0, g16);
#pragma unroll
      for (int idx = 0; idx < 16; ++idx) {
        const int ks = idx >> 3;
        const int dt = idx & 7;
        v16h bn;
        if (idx < 15) {
          const int ks2 = (idx + 1) >> 3;
          const int dt2 = (idx + 1) & 7;
          bn = frag_b(sVt + (dt2 * 16 + ln16) * SV_STR, 32 * ks2, g16);
        }
        acc[dt] = __builtin_amdgcn_wmma_f32_16x16x32_f16(false, aP[ks], false, bv,
                                                         (short)0, acc[dt], false, false);
        bv = bn;
      }
    }
  }

  // ---- Finalize: column sums of normalized rows (fused mean over L) ----
  float rinv[8];
#pragma unroll
  for (int r = 0; r < 8; ++r) rinv[r] = 1.0f / s_run[r];
#pragma unroll
  for (int dt = 0; dt < 8; ++dt) {
    float c = 0.0f;
#pragma unroll
    for (int r = 0; r < 8; ++r) c += acc[dt][r] * rinv[r];
    c += __shfl_xor(c, 16, 32);                 // combine half-wave row groups
    if (lane < 16) sRed[wv * 128 + dt * 16 + lane] = c;
  }
  __syncthreads();
  if (tid < 128) {
    float t = 0.0f;
#pragma unroll
    for (int w2 = 0; w2 < 8; ++w2) t += sRed[w2 * 128 + tid];
    ws[(size_t)wg * 128 + tid] = t;             // partial column sum for this q-tile
  }
}

// Reduce 16 query-tile partials per head, scale by 1/L (mean over query rows).
__global__ __launch_bounds__(128) void attn_reduce_kernel(const float* __restrict__ ws,
                                                          float* __restrict__ out) {
  const int head = blockIdx.x;
  const int d    = threadIdx.x;
  float t = 0.0f;
#pragma unroll
  for (int q = 0; q < 16; ++q) t += ws[((size_t)head * 16 + q) * 128 + d];
  out[head * 128 + d] = t * (1.0f / (float)L_SEQ);
}

extern "C" void kernel_launch(void* const* d_in, const int* in_sizes, int n_in,
                              void* d_out, int out_size, void* d_ws, size_t ws_size,
                              hipStream_t stream) {
  (void)in_sizes; (void)n_in; (void)out_size; (void)ws_size;
  const float* x  = (const float*)d_in[0];
  float* out      = (float*)d_out;
  float* ws       = (float*)d_ws;   // needs 64*16*128 floats = 512 KB

  const size_t smem =
      (size_t)(QT * SQ_STR + KT * SK_STR + D_DIM * SV_STR + QT * SP_STR) * sizeof(_Float16) +
      (size_t)8 * 128 * sizeof(float);  // ~91 KB, fine for 320 KB WGP LDS

  attn_partial_kernel<<<dim3(64 * 16), dim3(256), smem, stream>>>(x, ws);
  attn_reduce_kernel<<<dim3(64), dim3(128), 0, stream>>>(ws, out);
}
